// BaseGenerator_56100862820922
// MI455X (gfx1250) — matrix operand order repaired and checked by reading, not compiled, not run on hardware
//
#include <hip/hip_runtime.h>
#include <hip/hip_bf16.h>
#include <math.h>

typedef __bf16 bf16;
typedef __attribute__((ext_vector_type(16))) __bf16 v16bf;
typedef __attribute__((ext_vector_type(8)))  float   v8f;

#define BB   8
#define SS   512
#define EE   1024
#define HH   16
#define DDh  64
#define FFF  4096
#define LLn  6
#define VVo  100
#define MMr  (BB*SS)      /* 4096 rows */
#define NEGBIG (-1e9f)

union Frag { v16bf v; uint4 u[2]; };

__device__ __forceinline__ v8f wmma_bf16(v16bf a, v16bf b, v8f c) {
  return __builtin_amdgcn_wmma_f32_16x16x32_bf16(false, a, false, b, (short)0, c, false, false);
}

// A fragment: 16x32 (MxK) row-major tile, documented CDNA5 layout:
// lane = M (mod 16); half-wave selects K sub-blocks {0..7,16..23} vs {8..15,24..31}
__device__ __forceinline__ Frag load_fragA(const bf16* p, int ld) {
  int lane = threadIdx.x & 31;
  int r  = lane & 15;
  int kh = (lane >> 4) << 3;      // 0 or 8
  const bf16* q = p + r * ld + kh;
  Frag f;
  f.u[0] = *reinterpret_cast<const uint4*>(q);        // K = kh .. kh+7
  f.u[1] = *reinterpret_cast<const uint4*>(q + 16);   // K = kh+16 .. kh+23
  return f;
}

// B fragment: p points at B^T, row-major [N][K] (K-slice of 32).
// lane = N (mod 16); half-wave selects K sub-blocks {0..7,16..23} vs {8..15,24..31}
__device__ __forceinline__ Frag load_fragB(const bf16* p, int ld) {
  int lane = threadIdx.x & 31;
  int n  = lane & 15;
  int kb = (lane >> 4) << 3;      // 0 or 8
  const bf16* q = p + n * ld + kb;
  Frag f;
  f.u[0] = *reinterpret_cast<const uint4*>(q);        // K = kb .. kb+7
  f.u[1] = *reinterpret_cast<const uint4*>(q + 16);   // K = kb+16 .. kb+23
  return f;
}

// ---- CDNA5 async global->LDS copy (ASYNCcnt path, no VGPR staging) ----
__device__ __forceinline__ unsigned lds_off32(const void* p) {
  // flat-LDS aperture: addr[31:0] is the wave-relative LDS byte offset
  return (unsigned)(uintptr_t)p;
}
__device__ __forceinline__ void async_copy16(unsigned lds, const void* g) {
  asm volatile("global_load_async_to_lds_b128 %0, %1, off"
               :: "v"(lds), "v"((unsigned long long)(uintptr_t)g)
               : "memory");
}
__device__ __forceinline__ void wait_async0() {
  asm volatile("s_wait_asynccnt 0x0" ::: "memory");
}
__device__ __forceinline__ void wait_async2() {
  asm volatile("s_wait_asynccnt 0x2" ::: "memory");
}

// ---------------------------------------------------------------- utilities
__global__ void cvt_f32_bf16(const float* __restrict__ in, bf16* __restrict__ out, size_t n) {
  size_t i = (size_t)blockIdx.x * blockDim.x + threadIdx.x;
  if (i < n) out[i] = (bf16)in[i];
}

__global__ void embed_kernel(const int* __restrict__ seq, const float* __restrict__ tok,
                             float* __restrict__ xf, bf16* __restrict__ xb) {
  size_t i = (size_t)blockIdx.x * blockDim.x + threadIdx.x;   // over MMr*EE
  int e  = (int)(i & (EE - 1));
  int bs = (int)(i >> 10);
  float v = tok[(size_t)seq[bs] * EE + e] * 32.0f;            // sqrt(1024)
  xf[i] = v;
  xb[i] = (bf16)v;
}

// bias_pre[b,h,q,k] = dist+up+down+right gathers, causal & key-pad masked
__global__ void bias_kernel(const int* __restrict__ dsq, const int* __restrict__ usq,
                            const int* __restrict__ dnsq, const int* __restrict__ rsq,
                            const float* __restrict__ de, const float* __restrict__ ue,
                            const float* __restrict__ dne, const float* __restrict__ re,
                            const int* __restrict__ seq, float* __restrict__ out) {
  size_t idx = (size_t)blockIdx.x * blockDim.x + threadIdx.x; // over BB*SS*SS
  int k = (int)(idx % SS);
  size_t t = idx / SS;
  int q = (int)(t % SS);
  int b = (int)(t / SS);
  int di = dsq[idx] * HH, ui = usq[idx] * HH, wi = dnsq[idx] * HH, ri = rsq[idx] * HH;
  bool causal = (k <= q);
  float padadd = (seq[(size_t)b * SS + k] == 0) ? NEGBIG : 0.0f;
#pragma unroll
  for (int h = 0; h < HH; ++h) {
    float v = causal ? (de[di + h] + ue[ui + h] + dne[wi + h] + re[ri + h]) : NEGBIG;
    v += padadd;
    out[(((size_t)(b * HH + h) * SS + q) * SS) + k] = v;
  }
}

// ---------------------------------------------------------------- GEMM
// C[M,N] = A[M,K](bf16) @ W[N,K]^T(bf16) + bias ; optional GELU / residual
// Double-buffered LDS tiles filled by async global->LDS copies.
#define BM 128
#define BN 128
#define BK 32
#define LDT (BK + 8)   // padded LDS stride (elements)

__global__ __launch_bounds__(256) void gemm_kernel(
    const bf16* __restrict__ A, const bf16* __restrict__ W,
    const float* __restrict__ bias, const float* __restrict__ residual,
    float* __restrict__ outf, bf16* __restrict__ outb,
    int M, int N, int K, int do_gelu)
{
  __shared__ bf16 sA[2][BM * LDT];
  __shared__ bf16 sB[2][BN * LDT];
  int t = threadIdx.x;
  int wave = t >> 5, lane = t & 31;
  int wm = wave & 3, wn = wave >> 2;        // 4x2 wave grid -> 128x128
  int bm0 = blockIdx.y * BM;
  int bn0 = blockIdx.x * BN;

  v8f acc[2][4];
#pragma unroll
  for (int mt = 0; mt < 2; ++mt)
#pragma unroll
    for (int nt = 0; nt < 4; ++nt)
      acc[mt][nt] = (v8f){0.f,0.f,0.f,0.f,0.f,0.f,0.f,0.f};

  int lrow = t >> 1;              // 0..127
  int lcol = (t & 1) * 16;        // 0 / 16
  // clamp B row so every wave issues a uniform number of async ops
  // (rows >= N only feed output columns >= N which are never stored)
  int brow = bn0 + lrow; if (brow > N - 1) brow = N - 1;

  // stage issue: 2 async ops per wave (one A chunk, one B chunk per lane)
  auto issue = [&](int buf, int k0) {
    async_copy16(lds_off32(&sA[buf][lrow * LDT + lcol]),
                 A + (size_t)(bm0 + lrow) * K + k0 + lcol);
    async_copy16(lds_off32(&sB[buf][lrow * LDT + lcol]),
                 W + (size_t)brow * K + k0 + lcol);
  };

  int nstage = K / BK;
  issue(0, 0);
  for (int s = 0; s < nstage; ++s) {
    int cur = s & 1;
    if (s + 1 < nstage) { issue(cur ^ 1, (s + 1) * BK); wait_async2(); }
    else                { wait_async0(); }
    __syncthreads();

    const bf16* pA = sA[cur];
    const bf16* pB = sB[cur];
    Frag a0 = load_fragA(pA + (wm * 32)      * LDT, LDT);
    Frag a1 = load_fragA(pA + (wm * 32 + 16) * LDT, LDT);
    Frag bfr[4];
#pragma unroll
    for (int nt = 0; nt < 4; ++nt)
      bfr[nt] = load_fragB(pB + (wn * 64 + nt * 16) * LDT, LDT);
#pragma unroll
    for (int nt = 0; nt < 4; ++nt) {
      acc[0][nt] = wmma_bf16(a0.v, bfr[nt].v, acc[0][nt]);
      acc[1][nt] = wmma_bf16(a1.v, bfr[nt].v, acc[1][nt]);
    }
    __syncthreads();   // all reads of `cur` done before it is refilled
  }

  int mhalf = (lane >> 4) << 3, nl = lane & 15;
#pragma unroll
  for (int mt = 0; mt < 2; ++mt)
#pragma unroll
    for (int nt = 0; nt < 4; ++nt)
#pragma unroll
      for (int g = 0; g < 8; ++g) {
        int row = bm0 + wm * 32 + mt * 16 + g + mhalf;
        int col = bn0 + wn * 64 + nt * 16 + nl;
        if (col < N) {
          float v = acc[mt][nt][g] + (bias ? bias[col] : 0.0f);
          if (do_gelu) v = 0.5f * v * (1.0f + erff(v * 0.70710678118f));
          if (residual) v += residual[(size_t)row * N + col];
          if (outf) outf[(size_t)row * N + col] = v;
          if (outb) outb[(size_t)row * N + col] = (bf16)v;
        }
      }
}

// ---------------------------------------------------------------- attention
// Flash attention per (b,h,q-tile16); one wave per q-tile, 8 waves share K/V LDS tiles.
__global__ __launch_bounds__(256) void attn_kernel(
    const bf16* __restrict__ qkv, const float* __restrict__ biasp, bf16* __restrict__ out)
{
  const int LDK = 72, LDV = 40, LDP = 40;
  __shared__ bf16 sK[32 * 72];        // K block [32 keys][64 d]
  __shared__ bf16 sVt[64 * 40];       // V block transposed [64 d][32 keys]
  __shared__ bf16 sP[8][16 * 40];     // per-wave probabilities [16 q][32 keys]

  int h = blockIdx.y, b = blockIdx.z;
  int wave = threadIdx.x >> 5, lane = threadIdx.x & 31;
  int q0 = blockIdx.x * 128 + wave * 16;
  int mhalf = (lane >> 4) << 3, nl = lane & 15;

  // Q fragments (scaled by 1/sqrt(D) = 0.125)
  Frag qa[2];
  {
    const bf16* Qb = qkv + (size_t)(b * SS + q0) * (3 * EE) + h * DDh;
#pragma unroll
    for (int s = 0; s < 2; ++s) {
      qa[s] = load_fragA(Qb + s * 32, 3 * EE);
#pragma unroll
      for (int j = 0; j < 16; ++j)
        qa[s].v[j] = (bf16)((float)qa[s].v[j] * 0.125f);
    }
  }

  float rmax[8], rsum[8];
  v8f o[4];
#pragma unroll
  for (int g = 0; g < 8; ++g) { rmax[g] = -1e30f; rsum[g] = 0.0f; }
#pragma unroll
  for (int nt = 0; nt < 4; ++nt) o[nt] = (v8f){0.f,0.f,0.f,0.f,0.f,0.f,0.f,0.f};

  const float* bp = biasp + ((size_t)(b * HH + h) * SS + q0) * SS;

  for (int kb = 0; kb < SS; kb += 32) {
    __syncthreads();    // previous block's reads complete before refill
    { // K tile via async global->LDS; V loaded and transposed through VGPRs
      int r = threadIdx.x >> 3;             // key row 0..31
      int c = (threadIdx.x & 7) * 8;        // d offset 0..56
      const bf16* gk = qkv + (size_t)(b * SS + kb + r) * (3 * EE) + EE + h * DDh + c;
      async_copy16(lds_off32(&sK[r * LDK + c]), gk);
      const bf16* gv = qkv + (size_t)(b * SS + kb + r) * (3 * EE) + 2 * EE + h * DDh + c;
      uint4 vv = *reinterpret_cast<const uint4*>(gv);
      const bf16* pv = reinterpret_cast<const bf16*>(&vv);
#pragma unroll
      for (int j = 0; j < 8; ++j) sVt[(c + j) * LDV + r] = pv[j];
    }
    wait_async0();
    __syncthreads();

    // scores: two 16x16 tiles (keys kb..+15, kb+16..+31)
    v8f sc[2];
#pragma unroll
    for (int kt = 0; kt < 2; ++kt) {
      Frag k0f = load_fragB(sK + (kt * 16) * LDK + 0,  LDK);
      Frag k1f = load_fragB(sK + (kt * 16) * LDK + 32, LDK);
      v8f c = (v8f){0.f,0.f,0.f,0.f,0.f,0.f,0.f,0.f};
      c = wmma_bf16(qa[0].v, k0f.v, c);
      c = wmma_bf16(qa[1].v, k1f.v, c);
#pragma unroll
      for (int g = 0; g < 8; ++g)
        c[g] += bp[(size_t)(g + mhalf) * SS + kb + kt * 16 + nl];
      sc[kt] = c;
    }

    // online softmax (rows live in 16-lane half-waves)
    float nmax[8], scale[8], ps[8];
#pragma unroll
    for (int g = 0; g < 8; ++g) {
      float m2 = fmaxf(sc[0][g], sc[1][g]);
#pragma unroll
      for (int msk = 1; msk < 16; msk <<= 1) m2 = fmaxf(m2, __shfl_xor(m2, msk, 16));
      nmax[g] = fmaxf(rmax[g], m2);
      scale[g] = __expf(rmax[g] - nmax[g]);
      rmax[g] = nmax[g];
      ps[g] = 0.0f;
    }
#pragma unroll
    for (int kt = 0; kt < 2; ++kt)
#pragma unroll
      for (int g = 0; g < 8; ++g) {
        float p = __expf(sc[kt][g] - nmax[g]);
        sP[wave][(g + mhalf) * LDP + kt * 16 + nl] = (bf16)p;
        ps[g] += p;
      }
#pragma unroll
    for (int g = 0; g < 8; ++g) {
      float s2 = ps[g];
#pragma unroll
      for (int msk = 1; msk < 16; msk <<= 1) s2 += __shfl_xor(s2, msk, 16);
      rsum[g] = rsum[g] * scale[g] + s2;
    }
#pragma unroll
    for (int nt = 0; nt < 4; ++nt)
#pragma unroll
      for (int g = 0; g < 8; ++g) o[nt][g] *= scale[g];

    // P @ V  (same-wave LDS ops stay in order: store->load needs no barrier)
    Frag pf = load_fragA(&sP[wave][0], LDP);
#pragma unroll
    for (int nt = 0; nt < 4; ++nt) {
      Frag vf = load_fragB(sVt + (nt * 16) * LDV, LDV);
      o[nt] = wmma_bf16(pf.v, vf.v, o[nt]);
    }
  }

#pragma unroll
  for (int nt = 0; nt < 4; ++nt)
#pragma unroll
    for (int g = 0; g < 8; ++g) {
      float val = o[nt][g] / rsum[g];
      out[(size_t)(b * SS + q0 + g + mhalf) * EE + h * DDh + nt * 16 + nl] = (bf16)val;
    }
}

// ---------------------------------------------------------------- layernorm
__global__ __launch_bounds__(256) void ln_kernel(
    const float* __restrict__ in, const float* __restrict__ gam, const float* __restrict__ bet,
    float* __restrict__ outf, bf16* __restrict__ outb)
{
  __shared__ float red[256];
  int row = blockIdx.x, t = threadIdx.x;
  const float* p = in + (size_t)row * EE;
  float vals[4];
  float s = 0.0f;
#pragma unroll
  for (int i = 0; i < 4; ++i) { vals[i] = p[t + i * 256]; s += vals[i]; }
  red[t] = s; __syncthreads();
  for (int off = 128; off > 0; off >>= 1) { if (t < off) red[t] += red[t + off]; __syncthreads(); }
  float mean = red[0] * (1.0f / EE);
  __syncthreads();
  float s2 = 0.0f;
#pragma unroll
  for (int i = 0; i < 4; ++i) { float d = vals[i] - mean; s2 += d * d; }
  red[t] = s2; __syncthreads();
  for (int off = 128; off > 0; off >>= 1) { if (t < off) red[t] += red[t + off]; __syncthreads(); }
  float inv = rsqrtf(red[0] * (1.0f / EE) + 1e-5f);
#pragma unroll
  for (int i = 0; i < 4; ++i) {
    int idx = t + i * 256;
    float ov = (vals[i] - mean) * inv * gam[idx] + bet[idx];
    outf[(size_t)row * EE + idx] = ov;
    outb[(size_t)row * EE + idx] = (bf16)ov;
  }
}

// ---------------------------------------------------------------- launch
extern "C" void kernel_launch(void* const* d_in, const int* in_sizes, int n_in,
                              void* d_out, int out_size, void* d_ws, size_t ws_size,
                              hipStream_t stream) {
  (void)in_sizes; (void)n_in; (void)out_size; (void)ws_size;
  const int*   seq   = (const int*)d_in[0];
  const int*   dsq   = (const int*)d_in[1];
  const int*   usq   = (const int*)d_in[2];
  const int*   dnsq  = (const int*)d_in[3];
  const int*   rsq   = (const int*)d_in[4];
  const float* tok   = (const float*)d_in[5];
  const float* de    = (const float*)d_in[6];
  const float* ue    = (const float*)d_in[7];
  const float* dne   = (const float*)d_in[8];
  const float* re    = (const float*)d_in[9];
  const float* Wqkv  = (const float*)d_in[10];
  const float* bqkv  = (const float*)d_in[11];
  const float* Wo    = (const float*)d_in[12];
  const float* bo    = (const float*)d_in[13];
  const float* W1    = (const float*)d_in[14];
  const float* b1    = (const float*)d_in[15];
  const float* W2    = (const float*)d_in[16];
  const float* b2    = (const float*)d_in[17];
  const float* ln1g  = (const float*)d_in[18];
  const float* ln1b  = (const float*)d_in[19];
  const float* ln2g  = (const float*)d_in[20];
  const float* ln2b  = (const float*)d_in[21];
  const float* lnfg  = (const float*)d_in[22];
  const float* lnfb  = (const float*)d_in[23];
  const float* Wg    = (const float*)d_in[24];
  const float* bg    = (const float*)d_in[25];
  float* logits = (float*)d_out;

  // workspace carve-out
  size_t off = 0;
  char* base = (char*)d_ws;
  auto carve = [&](size_t bytes) -> void* {
    void* p = base + off;
    off += (bytes + 255) & ~(size_t)255;
    return p;
  };
  bf16*  wqkv_b = (bf16*)carve((size_t)LLn * 3 * EE * EE * 2);
  bf16*  wo_b   = (bf16*)carve((size_t)LLn * EE * EE * 2);
  bf16*  w1_b   = (bf16*)carve((size_t)LLn * FFF * EE * 2);
  bf16*  w2_b   = (bf16*)carve((size_t)LLn * EE * FFF * 2);
  bf16*  wg_b   = (bf16*)carve((size_t)VVo * EE * 2);
  float* biasp  = (float*)carve((size_t)BB * HH * SS * SS * 4);
  float* xf     = (float*)carve((size_t)MMr * EE * 4);
  bf16*  xb     = (bf16*) carve((size_t)MMr * EE * 2);
  bf16*  qkvb   = (bf16*) carve((size_t)MMr * 3 * EE * 2);
  bf16*  attb   = (bf16*) carve((size_t)MMr * EE * 2);
  float* resf   = (float*)carve((size_t)MMr * EE * 4);
  bf16*  hb     = (bf16*) carve((size_t)MMr * FFF * 2);

  auto cvt = [&](const float* src, bf16* dst, size_t n) {
    cvt_f32_bf16<<<(unsigned)((n + 255) / 256), 256, 0, stream>>>(src, dst, n);
  };
  cvt(Wqkv, wqkv_b, (size_t)LLn * 3 * EE * EE);
  cvt(Wo,   wo_b,   (size_t)LLn * EE * EE);
  cvt(W1,   w1_b,   (size_t)LLn * FFF * EE);
  cvt(W2,   w2_b,   (size_t)LLn * EE * FFF);
  cvt(Wg,   wg_b,   (size_t)VVo * EE);

  embed_kernel<<<(MMr * EE) / 256, 256, 0, stream>>>(seq, tok, xf, xb);
  bias_kernel<<<(BB * SS * SS) / 256, 256, 0, stream>>>(dsq, usq, dnsq, rsq,
                                                        de, ue, dne, re, seq, biasp);

  for (int l = 0; l < LLn; ++l) {
    const bf16*  wqkv_l = wqkv_b + (size_t)l * 3 * EE * EE;
    const float* bqkv_l = bqkv + (size_t)l * 3 * EE;
    const bf16*  wo_l   = wo_b + (size_t)l * EE * EE;
    const float* bo_l   = bo + (size_t)l * EE;
    const bf16*  w1_l   = w1_b + (size_t)l * FFF * EE;
    const float* b1_l   = b1 + (size_t)l * FFF;
    const bf16*  w2_l   = w2_b + (size_t)l * EE * FFF;
    const float* b2_l   = b2 + (size_t)l * EE;

    // QKV projection
    gemm_kernel<<<dim3(3 * EE / BN, MMr / BM), 256, 0, stream>>>(
        xb, wqkv_l, bqkv_l, nullptr, nullptr, qkvb, MMr, 3 * EE, EE, 0);
    // flash attention
    attn_kernel<<<dim3(SS / 128, HH, BB), 256, 0, stream>>>(qkvb, biasp, attb);
    // output projection + residual
    gemm_kernel<<<dim3(EE / BN, MMr / BM), 256, 0, stream>>>(
        attb, wo_l, bo_l, xf, resf, nullptr, MMr, EE, EE, 0);
    ln_kernel<<<MMr, 256, 0, stream>>>(resf, ln1g + (size_t)l * EE, ln1b + (size_t)l * EE, xf, xb);
    // FFN
    gemm_kernel<<<dim3(FFF / BN, MMr / BM), 256, 0, stream>>>(
        xb, w1_l, b1_l, nullptr, nullptr, hb, MMr, FFF, EE, 1);
    gemm_kernel<<<dim3(EE / BN, MMr / BM), 256, 0, stream>>>(
        hb, w2_l, b2_l, xf, resf, nullptr, MMr, EE, FFF, 0);
    ln_kernel<<<MMr, 256, 0, stream>>>(resf, ln2g + (size_t)l * EE, ln2b + (size_t)l * EE, xf, xb);
  }

  // final norm + logits
  ln_kernel<<<MMr, 256, 0, stream>>>(xf, lnfg, lnfb, resf, xb);
  gemm_kernel<<<dim3(1, MMr / BM), 256, 0, stream>>>(
      xb, wg_b, bg, nullptr, logits, nullptr, MMr, VVo, EE, 0);
}
